// LayeredWindowAttention_57878979280984
// MI455X (gfx1250) — compile-verified, matrix operand order checked by reading, not run on hardware
//
#include <hip/hip_runtime.h>
#include <hip/hip_bf16.h>

typedef __attribute__((ext_vector_type(16))) __bf16 v16bf;
typedef __attribute__((ext_vector_type(8)))  float  v8f;

#define TOK   98      // tokens per window
#define TP    112     // padded to 7 M-tiles of 16
#define CDIM  128
#define NHEAD 4
#define HD    32
#define NWIN  64
#define QSCALE 0.17677669529663687f   // 32^-0.5

// ---- LDS layout (bytes), all offsets/strides 16B-aligned ----
// S  : [112][116] f32          = 51968  (x bf16 staging [112][136] overlaps: 30464)
// QS : [4][112][40] bf16       = 35840
// KS : [4][112][40] bf16       = 35840
// VT : [4][32][128] bf16       = 32768   (v stored dim-major = transposed)
// P  : [112][136] bf16         = 30464
// O  : [112][136] bf16         = 30464
#define OFF_Q 51968
#define OFF_K 87808
#define OFF_VT 123648
#define OFF_P 156416
#define OFF_O 186880
#define SMEM_BYTES 217344

#define XSTR 136   // x / P / O row stride (bf16 elements)
#define QKSTR 40   // q / k row stride
#define VTSTR 128  // v-transposed row stride

__device__ inline __bf16 tobf(float x) { return (__bf16)x; }

union FragB { v16bf v; __bf16 e[16]; };
union FragC { v8f  v; float f[8]; };

// Fragment loader, element (rc, k) = base[rc*stride + k]; covers rc0..rc0+15, k0..k0+31.
// ISA 16-bit layout: lane holds rc = lane&15; kb = 8*(lane>=16); K = {kb..kb+7, 16+kb..16+kb+7}.
// With stride % 8 == 0 and k0 % 32 == 0, each half is one aligned 16-byte load.
__device__ inline v16bf frag_ld(const __bf16* __restrict__ base, int rc0, int k0,
                                int stride, int lane) {
  int rc = rc0 + (lane & 15);
  int kb = k0 + ((lane >> 4) << 3);
  const __bf16* p = base + rc * stride + kb;
  FragB f;
#pragma unroll
  for (int j = 0; j < 8; ++j) f.e[j] = p[j];
#pragma unroll
  for (int j = 0; j < 8; ++j) f.e[8 + j] = p[16 + j];
  return f.v;
}

__device__ inline v8f wmma_bf16(v16bf a, v16bf b, v8f c) {
  return __builtin_amdgcn_wmma_f32_16x16x32_bf16(false, a, false, b,
                                                 (short)0, c, false, false);
}

// ---- prep: convert weights f32 -> bf16 into workspace (L2-resident thereafter) ----
__global__ __launch_bounds__(256)
void convert_weights(const float* __restrict__ qkv_w,
                     const float* __restrict__ proj_w,
                     __bf16* __restrict__ wq, __bf16* __restrict__ wp) {
  int i = blockIdx.x * 256 + threadIdx.x;
  int stride = gridDim.x * 256;
  for (int j = i; j < 3 * CDIM * CDIM; j += stride) wq[j] = tobf(qkv_w[j]);
  for (int j = i; j < CDIM * CDIM; j += stride)     wp[j] = tobf(proj_w[j]);
}

__global__ __launch_bounds__(256)
void swin_window_attn(const float* __restrict__ x,
                      const float* __restrict__ mask,
                      const float* __restrict__ rpb,
                      const __bf16* __restrict__ wqkv,
                      const float* __restrict__ qkv_b,
                      const __bf16* __restrict__ wproj,
                      const float* __restrict__ proj_b,
                      const int*   __restrict__ rel_idx,
                      float* __restrict__ out) {
  extern __shared__ char smem[];
  float*  S  = (float*)smem;                    // [112][116] f32
  __bf16* QS = (__bf16*)(smem + OFF_Q);         // [4][112][40]
  __bf16* KS = (__bf16*)(smem + OFF_K);         // [4][112][40]
  __bf16* VT = (__bf16*)(smem + OFF_VT);        // [4][32][128] dim-major
  __bf16* P  = (__bf16*)(smem + OFF_P);         // [112][136]
  __bf16* O  = (__bf16*)(smem + OFF_O);         // [112][136]
  __bf16* XS = (__bf16*)smem;                   // x staging, overlaps S

  const int b    = blockIdx.x;
  const int tid  = threadIdx.x;
  const int lane = tid & 31;
  const int wv   = tid >> 5;          // 8 waves

  const float* xb = x    + (size_t)b * TOK * CDIM;
  const float* mb = mask + (size_t)(b % NWIN) * TOK * TOK;

  // ---- Phase 0: stage x as bf16 (float4 loads), zero paddings, zero VT ----
  {
    const float4* x4 = (const float4*)xb;                 // 98*128/4 = 3136
    for (int i = tid; i < TOK * CDIM / 4; i += 256) {
      int r = i >> 5, c = (i & 31) << 2;                  // 32 float4 per row
      float4 v = x4[i];
      __bf16* d = XS + r * XSTR + c;
      d[0] = tobf(v.x); d[1] = tobf(v.y); d[2] = tobf(v.z); d[3] = tobf(v.w);
    }
    for (int i = tid; i < TP * XSTR; i += 256) {
      int r = i / XSTR, c = i % XSTR;
      if (r >= TOK || c >= CDIM) XS[i] = (__bf16)0.f;
    }
    for (int i = tid; i < NHEAD * HD * VTSTR; i += 256) VT[i] = (__bf16)0.f;
  }
  __syncthreads();

  // ---- Phase 1: qkv = x @ qkv_w^T + qkv_b -> q(*scale)/k row-major, v dim-major ----
  for (int t = wv; t < 7 * 24; t += 8) {
    int mt = t % 7, nt = t / 7;
    int n  = nt * 16 + (lane & 15);
    FragC c;
    float bias = qkv_b[n];
#pragma unroll
    for (int r = 0; r < 8; ++r) c.f[r] = bias;
#pragma unroll
    for (int ks = 0; ks < 4; ++ks) {
      v16bf a  = frag_ld(XS, mt * 16, ks * 32, XSTR, lane);
      v16bf bm = frag_ld(wqkv, nt * 16, ks * 32, CDIM, lane);
      c.v = wmma_bf16(a, bm, c.v);
    }
    int which = n >> 7;             // 0=q 1=k 2=v (uniform per tile)
    int h     = (n >> 5) & 3;
    int d     = n & 31;
    if (which < 2) {
      __bf16* dst = (which == 0) ? QS : KS;
      float scl = (which == 0) ? QSCALE : 1.0f;
#pragma unroll
      for (int r = 0; r < 8; ++r) {
        int row = mt * 16 + r + ((lane >> 4) << 3);
        dst[(h * TP + row) * QKSTR + d] = tobf(c.f[r] * scl);
      }
    } else {
      // transposed store: lane owns dim d; 8 consecutive token rows -> 16B store
      int rowb = mt * 16 + ((lane >> 4) << 3);
      __bf16* dst = VT + (h * HD + d) * VTSTR + rowb;
#pragma unroll
      for (int r = 0; r < 8; ++r) dst[r] = tobf(c.f[r]);
    }
  }
  __syncthreads();

  // ---- Phase 2: per-head attention ----
  for (int h = 0; h < NHEAD; ++h) {
    const __bf16* qh = QS + h * TP * QKSTR;
    const __bf16* kh = KS + h * TP * QKSTR;
    const __bf16* vh = VT + h * HD * VTSTR;

    // 2a: S = q @ k^T   (K = 32, single WMMA per tile)
    for (int t = wv; t < 49; t += 8) {
      int mt = t / 7, nt = t % 7;
      FragC c;
#pragma unroll
      for (int r = 0; r < 8; ++r) c.f[r] = 0.f;
      v16bf a  = frag_ld(qh, mt * 16, 0, QKSTR, lane);
      v16bf bm = frag_ld(kh, nt * 16, 0, QKSTR, lane);   // B(n,k) = k[n][k]
      c.v = wmma_bf16(a, bm, c.v);
#pragma unroll
      for (int r = 0; r < 8; ++r) {
        int row = mt * 16 + r + ((lane >> 4) << 3);
        S[row * 116 + nt * 16 + (lane & 15)] = c.f[r];
      }
    }
    __syncthreads();

    // 2b: softmax rows (wave per row), fusing rpb bias + window mask
    for (int row = wv; row < TP; row += 8) {
      if (row < TOK) {
        float mx = -1e30f;
        for (int c = lane; c < TOK; c += 32) {
          float s = S[row * 116 + c]
                  + rpb[rel_idx[row * TOK + c] * NHEAD + h]
                  + mb[row * TOK + c];
          S[row * 116 + c] = s;
          mx = fmaxf(mx, s);
        }
#pragma unroll
        for (int i = 16; i > 0; i >>= 1) mx = fmaxf(mx, __shfl_xor(mx, i, 32));
        float sum = 0.f;
        for (int c = lane; c < TOK; c += 32) {
          float e = __expf(S[row * 116 + c] - mx);
          S[row * 116 + c] = e;
          sum += e;
        }
#pragma unroll
        for (int i = 16; i > 0; i >>= 1) sum += __shfl_xor(sum, i, 32);
        float inv = 1.0f / sum;
        for (int c = lane; c < CDIM; c += 32)
          P[row * XSTR + c] = (c < TOK) ? tobf(S[row * 116 + c] * inv)
                                        : (__bf16)0.f;
      } else {
        for (int c = lane; c < CDIM; c += 32) P[row * XSTR + c] = (__bf16)0.f;
      }
    }
    __syncthreads();

    // 2c: O[:, h*32 : h*32+32] = P @ v   (B from transposed v: contiguous loads)
    for (int t = wv; t < 14; t += 8) {
      int mt = t / 2, nt = t % 2;
      FragC c;
#pragma unroll
      for (int r = 0; r < 8; ++r) c.f[r] = 0.f;
#pragma unroll
      for (int ks = 0; ks < 4; ++ks) {
        v16bf a  = frag_ld(P, mt * 16, ks * 32, XSTR, lane);
        v16bf bm = frag_ld(vh, nt * 16, ks * 32, VTSTR, lane); // B(n,k)=vT[n][k]
        c.v = wmma_bf16(a, bm, c.v);
      }
#pragma unroll
      for (int r = 0; r < 8; ++r) {
        int row = mt * 16 + r + ((lane >> 4) << 3);
        O[row * XSTR + h * HD + nt * 16 + (lane & 15)] = tobf(c.f[r]);
      }
    }
    __syncthreads();
  }

  // ---- Phase 3: out = O @ proj_w^T + proj_b ----
  float* ob = out + (size_t)b * TOK * CDIM;
  for (int t = wv; t < 56; t += 8) {
    int mt = t % 7, nt = t / 7;
    int n  = nt * 16 + (lane & 15);
    FragC c;
    float bias = proj_b[n];
#pragma unroll
    for (int r = 0; r < 8; ++r) c.f[r] = bias;
#pragma unroll
    for (int ks = 0; ks < 4; ++ks) {
      v16bf a  = frag_ld(O, mt * 16, ks * 32, XSTR, lane);
      v16bf bm = frag_ld(wproj, nt * 16, ks * 32, CDIM, lane);
      c.v = wmma_bf16(a, bm, c.v);
    }
#pragma unroll
    for (int r = 0; r < 8; ++r) {
      int row = mt * 16 + r + ((lane >> 4) << 3);
      if (row < TOK) ob[(size_t)row * CDIM + n] = c.f[r];
    }
  }
}

extern "C" void kernel_launch(void* const* d_in, const int* in_sizes, int n_in,
                              void* d_out, int out_size, void* d_ws, size_t ws_size,
                              hipStream_t stream) {
  (void)in_sizes; (void)n_in; (void)ws_size; (void)out_size;
  const float* x      = (const float*)d_in[0];
  const float* mask   = (const float*)d_in[1];
  const float* rpb    = (const float*)d_in[2];
  const float* qkv_w  = (const float*)d_in[3];
  const float* qkv_b  = (const float*)d_in[4];
  const float* proj_w = (const float*)d_in[5];
  const float* proj_b = (const float*)d_in[6];
  const int*   rel    = (const int*)d_in[7];
  float* out = (float*)d_out;

  __bf16* wq = (__bf16*)d_ws;                         // 3*128*128 bf16
  __bf16* wp = wq + 3 * CDIM * CDIM;                  // 128*128 bf16

  convert_weights<<<dim3(64), dim3(256), 0, stream>>>(qkv_w, proj_w, wq, wp);
  swin_window_attn<<<dim3(4096), dim3(256), SMEM_BYTES, stream>>>(
      x, mask, rpb, wq, qkv_b, wp, proj_b, rel, out);
}
